// ITCA_55439437857390
// MI455X (gfx1250) — matrix operand order repaired and checked by reading, not compiled, not run on hardware
//
#include <hip/hip_runtime.h>
#include <hip/hip_bf16.h>

typedef __attribute__((ext_vector_type(16))) _Float16 v16h;
typedef __attribute__((ext_vector_type(8)))  _Float16 v8h;
typedef __attribute__((ext_vector_type(8)))  float    v8f;
typedef __attribute__((ext_vector_type(4)))  unsigned v4u;
typedef __attribute__((ext_vector_type(4)))  int      v4i;
typedef __attribute__((ext_vector_type(8)))  int      v8i;

// Problem constants
#define BB   8
#define CC   512
#define KK   2048
#define NN   2304   // 48*48
#define DD   128    // C/4

// GEMM tiling
#define TM   64
#define TN   128
#define KST  32
#define LDAP 40     // padded LDS row stride (halves) for A tile [TM x KST]
#define LDBP 136    // padded LDS row stride (halves) for Bt tile [KST x TN]
#define ASZ  (TM * LDAP)
#define BSZ  (KST * LDBP)

#ifdef __has_builtin
#  if __has_builtin(__builtin_amdgcn_tensor_load_to_lds) && __has_builtin(__builtin_amdgcn_s_wait_tensorcnt)
#    define USE_TDM 1
#  else
#    define USE_TDM 0
#  endif
#else
#  define USE_TDM 0
#endif

// ---------------------------------------------------------------------------
// Tensor Data Mover: async-copy one A tile [64 rows x 32 halves] global->LDS,
// with TDM-generated padding so LDS rows land at a 40-half stride (== LDAP).
// pad_interval code 3 -> every 16 DWORDs (one 64B tile row)
// pad_amount   code 3 -> 4 DWORDs (8 halves) of pad  => 40-half row stride
// ---------------------------------------------------------------------------
#if USE_TDM
__device__ __forceinline__ void tdm_load_a(const _Float16* gaddr, unsigned lds_off, int kd) {
  unsigned long long ga = (unsigned long long)(uintptr_t)gaddr;
  v4u g0;
  g0[0] = 1u;                                            // count = 1 valid descriptor
  g0[1] = lds_off;                                       // lds_addr
  g0[2] = (unsigned)ga;                                  // global_addr[31:0]
  g0[3] = (unsigned)((ga >> 32) & 0x1FFFFFFu) | (2u << 30); // global_addr[56:32] | type=2
  v8i g1;
  g1[0] = (int)((1u << 16) | (1u << 20) | (3u << 22) | (3u << 25)); // data_size=2B, pad en/intv/amt
  g1[1] = (int)(((unsigned)kd & 0xFFFFu) << 16);         // tensor_dim0 lo16 in [63:48]
  g1[2] = (int)(((unsigned)kd >> 16) & 0xFFFFu);         // tensor_dim0 hi16; tensor_dim1 lo16 = 0
  g1[3] = (int)(0x10u | (32u << 16));                    // tensor_dim1 hi (2^20 rows), tile_dim0=32
  g1[4] = 64;                                            // tile_dim1=64, tile_dim2=0
  g1[5] = kd;                                            // tensor_dim0_stride lo32 (= lda)
  g1[6] = 0;
  g1[7] = 0;
  v4i z4 = {};
#if defined(__clang_major__) && (__clang_major__ >= 23)
  v8i z8 = {};
  __builtin_amdgcn_tensor_load_to_lds(g0, g1, z4, z4, z8, 0);
#else
  __builtin_amdgcn_tensor_load_to_lds(g0, g1, z4, z4, 0);
#endif
}
#endif

// ---------------------------------------------------------------------------
// Manual LDS staging (A fallback + B transpose)
// ---------------------------------------------------------------------------
__device__ __forceinline__ void stageA_manual(const _Float16* __restrict__ Ag, size_t lda,
                                              int m0, int k0, _Float16* As) {
  int t   = threadIdx.x;       // 0..255
  int row = t >> 2;            // 0..63
  int ch  = (t & 3) << 3;      // 0,8,16,24
  const _Float16* src = Ag + (size_t)(m0 + row) * lda + (k0 + ch);
  v8h v = *(const v8h*)src;
  *(v8h*)(As + row * LDAP + ch) = v;
}

__device__ __forceinline__ void stageB(const _Float16* __restrict__ Bg, size_t ldb,
                                       int n0, int k0, _Float16* Bt) {
#pragma unroll
  for (int i = 0; i < 2; ++i) {
    int idx = threadIdx.x + i * 256;  // 0..511
    int row = idx >> 2;               // n' 0..127
    int ch  = (idx & 3) << 3;         // k-chunk
    const _Float16* src = Bg + (size_t)(n0 + row) * ldb + (k0 + ch);
    __builtin_prefetch(src + KST, 0, 0);
    v8h v = *(const v8h*)src;
#pragma unroll
    for (int j = 0; j < 8; ++j) Bt[(ch + j) * LDBP + row] = v[j];  // transpose into LDS
  }
}

template <bool DUAL>
__device__ __forceinline__ void stageStep(const _Float16* __restrict__ Ag, size_t lda,
                                          const _Float16* __restrict__ B0,
                                          const _Float16* __restrict__ B1, size_t ldb,
                                          int m0, int n0, int k0, int wave,
                                          _Float16* AsBuf, _Float16* Bt0Buf, _Float16* Bt1Buf) {
#if USE_TDM
  if (wave == 0)
    tdm_load_a(Ag + (size_t)m0 * lda + k0, (unsigned)(uintptr_t)AsBuf, (int)lda);
#else
  stageA_manual(Ag, lda, m0, k0, AsBuf);
#endif
  stageB(B0, ldb, n0, k0, Bt0Buf);
  if (DUAL) stageB(B1, ldb, n0, k0, Bt1Buf);
}

// ---------------------------------------------------------------------------
// One k-step of compute: batch all fragment loads, then a WMMA clause.
// ---------------------------------------------------------------------------
template <bool DUAL>
__device__ __forceinline__ void computeStep(const _Float16* As, const _Float16* Bt0,
                                            const _Float16* Bt1, int lane, int mbase,
                                            int nreg, v8f acc0[4], v8f acc1[4]) {
  // A fragment per ISA: lanes 0-15 = M rows, k-halves {koff..+7, koff+16..+23}
  v16h a;
  {
    int m  = mbase + (lane & 15);
    int ko = (lane >> 4) << 3;
    v8h lo = *(const v8h*)(As + m * LDAP + ko);
    v8h hi = *(const v8h*)(As + m * LDAP + ko + 16);
#pragma unroll
    for (int i = 0; i < 8; ++i) { a[i] = lo[i]; a[8 + i] = hi[i]; }
  }
  // B fragments per ISA: lanes 0-15 = N cols, k-halves {koff..+7, koff+16..+23}
  v16h b0f[4];
  v16h b1f[4];
#pragma unroll
  for (int j = 0; j < 4; ++j) {
    int nb = nreg + j * 16 + (lane & 15);
    int kb = (lane >> 4) << 3;
#pragma unroll
    for (int i = 0; i < 8; ++i) { b0f[j][i] = Bt0[(kb + i) * LDBP + nb]; b0f[j][8 + i] = Bt0[(kb + 16 + i) * LDBP + nb]; }
    if (DUAL) {
#pragma unroll
      for (int i = 0; i < 8; ++i) { b1f[j][i] = Bt1[(kb + i) * LDBP + nb]; b1f[j][8 + i] = Bt1[(kb + 16 + i) * LDBP + nb]; }
    }
  }
#pragma unroll
  for (int j = 0; j < 4; ++j)
    acc0[j] = __builtin_amdgcn_wmma_f32_16x16x32_f16(false, a, false, b0f[j], (short)0,
                                                     acc0[j], false, false);
  if (DUAL) {
#pragma unroll
    for (int j = 0; j < 4; ++j)
      acc1[j] = __builtin_amdgcn_wmma_f32_16x16x32_f16(false, a, false, b1f[j], (short)0,
                                                       acc1[j], false, false);
  }
}

// ---------------------------------------------------------------------------
// Core NT GEMM, double-buffered: block tile 64(M) x 128(N'), 8 waves,
// each wave 16x64 (4 wmma accumulators per B operand).
// ---------------------------------------------------------------------------
template <bool DUAL>
__device__ __forceinline__ void gemmCore(const _Float16* __restrict__ Ag, size_t lda,
                                         const _Float16* __restrict__ B0,
                                         const _Float16* __restrict__ B1, size_t ldb,
                                         int Kd, int m0, int n0,
                                         _Float16* As, _Float16* Bt0, _Float16* Bt1,
                                         v8f acc0[4], v8f acc1[4]) {
  const int lane  = threadIdx.x & 31;
  const int wave  = threadIdx.x >> 5;
  const int mbase = (wave & 3) * 16;
  const int nreg  = (wave >> 2) * 64;

#pragma unroll
  for (int j = 0; j < 4; ++j)
#pragma unroll
    for (int r = 0; r < 8; ++r) { acc0[j][r] = 0.f; if (DUAL) acc1[j][r] = 0.f; }

  stageStep<DUAL>(Ag, lda, B0, B1, ldb, m0, n0, 0, wave, As, Bt0, Bt1);
#if USE_TDM
  if (wave == 0) __builtin_amdgcn_s_wait_tensorcnt(0);
#endif
  __syncthreads();

  int it = 0;
  for (int k0 = 0; k0 < Kd; k0 += KST, ++it) {
    const int cur = it & 1;
    if (k0 + KST < Kd)
      stageStep<DUAL>(Ag, lda, B0, B1, ldb, m0, n0, k0 + KST, wave,
                      As + (cur ^ 1) * ASZ, Bt0 + (cur ^ 1) * BSZ,
                      DUAL ? (Bt1 + (cur ^ 1) * BSZ) : nullptr);
    computeStep<DUAL>(As + cur * ASZ, Bt0 + cur * BSZ,
                      DUAL ? (Bt1 + cur * BSZ) : nullptr, lane, mbase, nreg, acc0, acc1);
#if USE_TDM
    if (wave == 0) __builtin_amdgcn_s_wait_tensorcnt(0);
#endif
    __syncthreads();
  }
}

// C/D layout helpers: lane<16 -> (M=r, N=lane); lane>=16 -> (M=r+8, N=lane-16)
#define EPI_IDX()                                        \
  const int lane = threadIdx.x & 31;                     \
  const int wave = threadIdx.x >> 5;                     \
  const int gm   = (int)blockIdx.x * TM + (wave & 3) * 16 + ((lane >> 4) << 3); \
  const int colb = (int)blockIdx.y * TN + (wave >> 2) * 64 + (lane & 15);

// ---------------------------------------------------------------------------
// Elementwise f32 -> f16
// ---------------------------------------------------------------------------
__global__ void k_cvt_h(const float* __restrict__ in, _Float16* __restrict__ out, size_t n) {
  size_t i = (size_t)blockIdx.x * 256 + threadIdx.x;
  size_t st = (size_t)gridDim.x * 256;
  for (; i < n; i += st) out[i] = (_Float16)in[i];
}

// ---------------------------------------------------------------------------
// q projection (tiny: ~1% of FLOPs) -> qh[b][n][128] f16
// ---------------------------------------------------------------------------
__global__ __launch_bounds__(256) void k_qproj(const float* __restrict__ x1,
                                               const float* __restrict__ x2,
                                               const float* __restrict__ wq1,
                                               const float* __restrict__ bq1,
                                               const float* __restrict__ wq2,
                                               const float* __restrict__ bq2,
                                               _Float16* __restrict__ qh) {
  int n = blockIdx.x * 256 + threadIdx.x;
  int o = blockIdx.y;  // 0..127
  int b = blockIdx.z;
  const float* x = (o < 64) ? x1 : x2;
  const float* w = (o < 64) ? (wq1 + (size_t)o * CC) : (wq2 + (size_t)(o - 64) * CC);
  float acc = (o < 64) ? bq1[o] : bq2[o - 64];
  const float* xp = x + (size_t)b * CC * NN + n;
#pragma unroll 4
  for (int c = 0; c < CC; ++c) acc += xp[(size_t)c * NN] * w[c];
  qh[((size_t)b * NN + n) * DD + o] = (_Float16)acc;
}

// ---------------------------------------------------------------------------
// key projection: key[b][k][128] = yh@wkh^T + bk   (normal row-major store)
// ---------------------------------------------------------------------------
__global__ __launch_bounds__(256) void k_keyproj(const _Float16* __restrict__ yh,
                                                 const _Float16* __restrict__ wkh,
                                                 const float* __restrict__ bk,
                                                 _Float16* __restrict__ keyh) {
  __shared__ _Float16 As[2 * ASZ];
  __shared__ _Float16 Bt[2 * BSZ];
  int b = blockIdx.z;
  v8f acc[4], dmy[4];
  gemmCore<false>(yh + (size_t)b * KK * CC, CC, wkh, nullptr, CC, CC,
                  (int)blockIdx.x * TM, (int)blockIdx.y * TN, As, Bt, nullptr, acc, dmy);
  EPI_IDX();
  _Float16* out = keyh + (size_t)b * KK * DD;
#pragma unroll
  for (int j = 0; j < 4; ++j) {
    int col = colb + j * 16;
    float bias = bk[col];
#pragma unroll
    for (int r = 0; r < 8; ++r)
      out[(size_t)(gm + r) * DD + col] = (_Float16)(acc[j][r] + bias);
  }
}

// ---------------------------------------------------------------------------
// val projection, stored TRANSPOSED: valT[b][c][k] f16 (so att@val is NT form)
// ---------------------------------------------------------------------------
__global__ __launch_bounds__(256) void k_valproj(const _Float16* __restrict__ yh,
                                                 const _Float16* __restrict__ wvh,
                                                 const float* __restrict__ bv,
                                                 _Float16* __restrict__ valT) {
  __shared__ _Float16 As[2 * ASZ];
  __shared__ _Float16 Bt[2 * BSZ];
  int b = blockIdx.z;
  v8f acc[4], dmy[4];
  gemmCore<false>(yh + (size_t)b * KK * CC, CC, wvh, nullptr, CC, CC,
                  (int)blockIdx.x * TM, (int)blockIdx.y * TN, As, Bt, nullptr, acc, dmy);
  EPI_IDX();
#pragma unroll
  for (int j = 0; j < 4; ++j) {
    int col = colb + j * 16;
    float bias = bv[col];
    v8h pk;
#pragma unroll
    for (int r = 0; r < 8; ++r) pk[r] = (_Float16)(acc[j][r] + bias);
    *(v8h*)(valT + ((size_t)b * CC + col) * KK + gm) = pk;  // contiguous 16B store
  }
}

// ---------------------------------------------------------------------------
// energy = |q@key1^T - q@key2^T|  -> e[b][n][k] f32
// ---------------------------------------------------------------------------
__global__ __launch_bounds__(256) void k_energy(const _Float16* __restrict__ qh,
                                                const _Float16* __restrict__ kh1,
                                                const _Float16* __restrict__ kh2,
                                                float* __restrict__ e) {
  __shared__ _Float16 As[2 * ASZ];
  __shared__ _Float16 Bt0[2 * BSZ];
  __shared__ _Float16 Bt1[2 * BSZ];
  int b = blockIdx.z;
  v8f a0[4], a1[4];
  gemmCore<true>(qh + (size_t)b * NN * DD, DD,
                 kh1 + (size_t)b * KK * DD, kh2 + (size_t)b * KK * DD, DD, DD,
                 (int)blockIdx.x * TM, (int)blockIdx.y * TN, As, Bt0, Bt1, a0, a1);
  EPI_IDX();
  float* eb = e + (size_t)b * NN * KK;
#pragma unroll
  for (int j = 0; j < 4; ++j) {
    int col = colb + j * 16;
#pragma unroll
    for (int r = 0; r < 8; ++r)
      eb[(size_t)(gm + r) * KK + col] = fabsf(a0[j][r] - a1[j][r]);
  }
}

// ---------------------------------------------------------------------------
// denom[b][k] = max(sum_n e, EPS); global min/max of e/denom (ordered-uint atomics,
// valid since all values >= 0)
// ---------------------------------------------------------------------------
__global__ __launch_bounds__(256) void k_denom(const float* __restrict__ e,
                                               float* __restrict__ denom,
                                               unsigned* __restrict__ gminb,
                                               unsigned* __restrict__ gmaxb) {
  int k = blockIdx.x * 256 + threadIdx.x;
  int b = blockIdx.y;
  const float* Eb = e + (size_t)b * NN * KK + k;
  float s = 0.f;
  for (int n = 0; n < NN; ++n) s += Eb[(size_t)n * KK];
  float d = fmaxf(s, 1e-12f);
  denom[b * KK + k] = d;
  float inv = 1.f / d;
  float mn = 3.4e38f, mx = 0.f;
  for (int n = 0; n < NN; ++n) {
    float v = Eb[(size_t)n * KK] * inv;
    mn = fminf(mn, v);
    mx = fmaxf(mx, v);
  }
  atomicMin(gminb, __float_as_uint(mn));
  atomicMax(gmaxb, __float_as_uint(mx));
}

// ---------------------------------------------------------------------------
// per-row softmax -> att f16; cmask/inv f16; per-batch counts
// ---------------------------------------------------------------------------
__global__ __launch_bounds__(256) void k_att(const float* __restrict__ e,
                                             const float* __restrict__ denom,
                                             const unsigned* __restrict__ gminb,
                                             const unsigned* __restrict__ gmaxb,
                                             _Float16* __restrict__ atth,
                                             _Float16* __restrict__ invh,
                                             float* __restrict__ ncb,
                                             float* __restrict__ csum) {
  __shared__ float sred[256];
  int row = blockIdx.x, b = blockIdx.y, t = threadIdx.x;
  const float* Er = e + ((size_t)b * NN + row) * KK;

  float m = -3.4e38f;
  for (int k = t; k < KK; k += 256) m = fmaxf(m, Er[k]);
  sred[t] = m; __syncthreads();
  for (int o = 128; o > 0; o >>= 1) { if (t < o) sred[t] = fmaxf(sred[t], sred[t + o]); __syncthreads(); }
  m = sred[0]; __syncthreads();

  float s = 0.f;
  for (int k = t; k < KK; k += 256) s += __expf(Er[k] - m);
  sred[t] = s; __syncthreads();
  for (int o = 128; o > 0; o >>= 1) { if (t < o) sred[t] += sred[t + o]; __syncthreads(); }
  s = sred[0]; __syncthreads();
  float invs = 1.f / s;

  float gmin = __uint_as_float(*gminb), gmax = __uint_as_float(*gmaxb);
  float rng = gmax - gmin;
  float invr = (rng > 0.f) ? 1.f / rng : 0.f;
  const float* dn = denom + b * KK;
  _Float16* ap = atth + ((size_t)b * NN + row) * KK;
  _Float16* ip = invh + ((size_t)b * NN + row) * KK;

  float c0 = 0.f, cs = 0.f;
  for (int k = t; k < KK; k += 256) {
    float ev = Er[k];
    ap[k] = (_Float16)(__expf(ev - m) * invs);
    float nv = (ev / dn[k] - gmin) * invr;
    float cm = (nv > 0.3f) ? 1.f : ((nv < 0.3f) ? 0.f : 0.5f);
    ip[k] = (_Float16)(1.f - cm);
    if (cm == 0.f) c0 += 1.f;
    cs += cm;
  }
  sred[t] = c0; __syncthreads();
  for (int o = 128; o > 0; o >>= 1) { if (t < o) sred[t] += sred[t + o]; __syncthreads(); }
  if (t == 0) atomicAdd(&ncb[b], sred[0]);
  __syncthreads();
  sred[t] = cs; __syncthreads();
  for (int o = 128; o > 0; o >>= 1) { if (t < o) sred[t] += sred[t + o]; __syncthreads(); }
  if (t == 0) atomicAdd(csum, sred[0]);
}

// ---------------------------------------------------------------------------
// out{1,2}[b][c][n] = scale*(att@val)^T + x    (dual-B WMMA, shared A fragments)
// ---------------------------------------------------------------------------
__global__ __launch_bounds__(256) void k_out(const _Float16* __restrict__ atth,
                                             const _Float16* __restrict__ vt1,
                                             const _Float16* __restrict__ vt2,
                                             const float* __restrict__ x1,
                                             const float* __restrict__ x2,
                                             const float* __restrict__ scalep,
                                             float* __restrict__ out1,
                                             float* __restrict__ out2) {
  __shared__ _Float16 As[2 * ASZ];
  __shared__ _Float16 Bt0[2 * BSZ];
  __shared__ _Float16 Bt1[2 * BSZ];
  int b = blockIdx.z;
  v8f a0[4], a1[4];
  gemmCore<true>(atth + (size_t)b * NN * KK, KK,
                 vt1 + (size_t)b * CC * KK, vt2 + (size_t)b * CC * KK, KK, KK,
                 (int)blockIdx.x * TM, (int)blockIdx.y * TN, As, Bt0, Bt1, a0, a1);
  EPI_IDX();
  float sc = scalep[0];
#pragma unroll
  for (int j = 0; j < 4; ++j) {
    int col = colb + j * 16;
    size_t base = ((size_t)b * CC + col) * NN + gm;
#pragma unroll
    for (int r = 0; r < 8; ++r) {
      out1[base + r] = sc * a0[j][r] + x1[base + r];
      out2[base + r] = sc * a1[j][r] + x2[base + r];
    }
  }
}

// ---------------------------------------------------------------------------
// res reduction: sum |inv@val1 - inv@val2| per batch
// ---------------------------------------------------------------------------
__global__ __launch_bounds__(256) void k_res(const _Float16* __restrict__ invh,
                                             const _Float16* __restrict__ vt1,
                                             const _Float16* __restrict__ vt2,
                                             float* __restrict__ resab) {
  __shared__ _Float16 As[2 * ASZ];
  __shared__ _Float16 Bt0[2 * BSZ];
  __shared__ _Float16 Bt1[2 * BSZ];
  __shared__ float red;
  int b = blockIdx.z;
  v8f a0[4], a1[4];
  gemmCore<true>(invh + (size_t)b * NN * KK, KK,
                 vt1 + (size_t)b * CC * KK, vt2 + (size_t)b * CC * KK, KK, KK,
                 (int)blockIdx.x * TM, (int)blockIdx.y * TN, As, Bt0, Bt1, a0, a1);
  float local = 0.f;
#pragma unroll
  for (int j = 0; j < 4; ++j)
#pragma unroll
    for (int r = 0; r < 8; ++r) local += fabsf(a0[j][r] - a1[j][r]);
  if (threadIdx.x == 0) red = 0.f;
  __syncthreads();
  atomicAdd(&red, local);
  __syncthreads();
  if (threadIdx.x == 0) atomicAdd(&resab[b], red);
}

// ---------------------------------------------------------------------------
// init / finalize
// ---------------------------------------------------------------------------
__global__ void k_init(unsigned* gminb, unsigned* gmaxb, float* ncb, float* resab, float* csum) {
  int t = threadIdx.x;
  if (t == 0) { *gminb = 0x7f800000u; *gmaxb = 0u; *csum = 0.f; }
  if (t < BB) { ncb[t] = 0.f; resab[t] = 0.f; }
}

__global__ void k_fin(const float* ncb, const float* resab, const float* csum, float* outp) {
  if (threadIdx.x == 0) {
    float la = *csum / ((float)BB * (float)NN * (float)KK);
    float rl = 0.f;
    for (int b = 0; b < BB; ++b)
      rl += (resab[b] / ((float)CC * (float)NN)) * ((float)NN * (float)KK) / (ncb[b] + 1.f);
    rl *= (1.f / (float)BB);
    outp[0] = rl;
    outp[1] = la;
  }
}

// ---------------------------------------------------------------------------
// launch
// ---------------------------------------------------------------------------
extern "C" void kernel_launch(void* const* d_in, const int* in_sizes, int n_in,
                              void* d_out, int out_size, void* d_ws, size_t ws_size,
                              hipStream_t stream) {
  (void)in_sizes; (void)n_in; (void)out_size; (void)ws_size;
  const float* x1  = (const float*)d_in[0];
  const float* y1  = (const float*)d_in[1];
  const float* x2  = (const float*)d_in[2];
  const float* y2  = (const float*)d_in[3];
  const float* wq1 = (const float*)d_in[4];
  const float* bq1 = (const float*)d_in[5];
  const float* wq2 = (const float*)d_in[6];
  const float* bq2 = (const float*)d_in[7];
  const float* wk1 = (const float*)d_in[8];
  const float* bk1 = (const float*)d_in[9];
  const float* wk2 = (const float*)d_in[10];
  const float* bk2 = (const float*)d_in[11];
  const float* wv1 = (const float*)d_in[12];
  const float* bv1 = (const float*)d_in[13];
  const float* wv2 = (const float*)d_in[14];
  const float* bv2 = (const float*)d_in[15];
  const float* scl = (const float*)d_in[16];

  const size_t BNK = (size_t)BB * NN * KK;   // 37,748,736
  const size_t BKC = (size_t)BB * KK * CC;   // 8,388,608
  const size_t BCN = (size_t)BB * CC * NN;   // 9,437,184

  char* ws = (char*)d_ws;
  size_t off = 0;
  auto alloc = [&](size_t bytes) -> char* {
    char* p = ws + off;
    off = (off + bytes + 255) & ~(size_t)255;
    return p;
  };

  float*     E    = (float*)alloc(BNK * 4);
  _Float16*  ATT  = (_Float16*)alloc(BNK * 2);
  _Float16*  INV  = (_Float16*)alloc(BNK * 2);
  _Float16*  YH1  = (_Float16*)alloc(BKC * 2);
  _Float16*  YH2  = (_Float16*)alloc(BKC * 2);
  _Float16*  KH1  = (_Float16*)alloc((size_t)BB * KK * DD * 2);
  _Float16*  KH2  = (_Float16*)alloc((size_t)BB * KK * DD * 2);
  _Float16*  VT1  = (_Float16*)alloc((size_t)BB * CC * KK * 2);
  _Float16*  VT2  = (_Float16*)alloc((size_t)BB * CC * KK * 2);
  _Float16*  QH   = (_Float16*)alloc((size_t)BB * NN * DD * 2);
  _Float16*  WKH1 = (_Float16*)alloc((size_t)DD * CC * 2);
  _Float16*  WKH2 = (_Float16*)alloc((size_t)DD * CC * 2);
  _Float16*  WVH1 = (_Float16*)alloc((size_t)CC * CC * 2);
  _Float16*  WVH2 = (_Float16*)alloc((size_t)CC * CC * 2);
  float*     DEN  = (float*)alloc((size_t)BB * KK * 4);
  unsigned*  GMIN = (unsigned*)alloc(4);
  unsigned*  GMAX = (unsigned*)alloc(4);
  float*     CSUM = (float*)alloc(4);
  float*     NC   = (float*)alloc(BB * 4);
  float*     RES  = (float*)alloc(BB * 4);

  float* out1 = (float*)d_out;
  float* out2 = out1 + BCN;
  float* lossp = out1 + 2 * BCN;

  k_init<<<1, 32, 0, stream>>>(GMIN, GMAX, NC, RES, CSUM);

  k_cvt_h<<<4096, 256, 0, stream>>>(y1, YH1, BKC);
  k_cvt_h<<<4096, 256, 0, stream>>>(y2, YH2, BKC);
  k_cvt_h<<<256, 256, 0, stream>>>(wk1, WKH1, (size_t)DD * CC);
  k_cvt_h<<<256, 256, 0, stream>>>(wk2, WKH2, (size_t)DD * CC);
  k_cvt_h<<<1024, 256, 0, stream>>>(wv1, WVH1, (size_t)CC * CC);
  k_cvt_h<<<1024, 256, 0, stream>>>(wv2, WVH2, (size_t)CC * CC);

  k_qproj<<<dim3(NN / 256, DD, BB), 256, 0, stream>>>(x1, x2, wq1, bq1, wq2, bq2, QH);

  k_keyproj<<<dim3(KK / TM, DD / TN, BB), 256, 0, stream>>>(YH1, WKH1, bk1, KH1);
  k_keyproj<<<dim3(KK / TM, DD / TN, BB), 256, 0, stream>>>(YH2, WKH2, bk2, KH2);
  k_valproj<<<dim3(KK / TM, CC / TN, BB), 256, 0, stream>>>(YH1, WVH1, bv1, VT1);
  k_valproj<<<dim3(KK / TM, CC / TN, BB), 256, 0, stream>>>(YH2, WVH2, bv2, VT2);

  k_energy<<<dim3(NN / TM, KK / TN, BB), 256, 0, stream>>>(QH, KH1, KH2, E);

  k_denom<<<dim3(KK / 256, BB), 256, 0, stream>>>(E, DEN, GMIN, GMAX);
  k_att<<<dim3(NN, BB), 256, 0, stream>>>(E, DEN, GMIN, GMAX, ATT, INV, NC, CSUM);

  k_out<<<dim3(NN / TM, CC / TN, BB), 256, 0, stream>>>(ATT, VT1, VT2, x1, x2, scl, out1, out2);
  k_res<<<dim3(NN / TM, CC / TN, BB), 256, 0, stream>>>(INV, VT1, VT2, RES);

  k_fin<<<1, 32, 0, stream>>>(NC, RES, CSUM, lossp);
}
